// CharRNN_63582695850682
// MI455X (gfx1250) — matrix operand order repaired and checked by reading, not compiled, not run on hardware
//
#include <hip/hip_runtime.h>
#include <hip/hip_bf16.h>

// ---- problem dims ----
#define B_    32
#define T_    128
#define H_    1024
#define V_    32000
#define FOURH 4096
#define ROWS  (B_ * T_)      // 4096
#define EPS_  1e-5f
#define KC    32             // K depth of one WMMA
#define NBLK  64             // blocks in persistent recurrent kernel

typedef __attribute__((ext_vector_type(16))) _Float16 v16h;
typedef __attribute__((ext_vector_type(8)))  float    v8f;

union Frag16 { v16h v; uint4 q[2]; _Float16 h[16]; };

// A 16x32 f16 frag (MxK) from row-major activations:
// lanes 0-15 -> M=lane, K = 0..7 & 16..23 ; lanes 16-31 -> M=lane-16, K = 8..15 & 24..31
__device__ inline v16h load_a_frag(const _Float16* __restrict__ A, int lda) {
    int lane = threadIdx.x & 31;
    int m    = lane & 15;
    int kb   = (lane >> 4) << 3;            // 0 or 8
    const _Float16* p = A + (size_t)m * lda + kb;
    Frag16 r;
    r.q[0] = *reinterpret_cast<const uint4*>(p);
    r.q[1] = *reinterpret_cast<const uint4*>(p + 16);
    return r.v;
}

// B 32x16 frag from PRE-TRANSPOSED weights Bt[N][K] (row n = one output column):
// lanes 0-15 -> N=lane, K = 0..7 & 16..23 ; lanes 16-31 -> N=lane-16, K = 8..15 & 24..31
__device__ inline v16h load_bT_frag(const _Float16* __restrict__ Bt, int ldk) {
    int lane = threadIdx.x & 31;
    int n    = lane & 15;
    int kb   = (lane >> 4) << 3;            // 0 or 8
    const _Float16* p = Bt + (size_t)n * ldk + kb;
    Frag16 r;
    r.q[0] = *reinterpret_cast<const uint4*>(p);
    r.q[1] = *reinterpret_cast<const uint4*>(p + 16);
    return r.v;
}

// ---------------- reductions (wave32) ----------------
__device__ inline float block_sum(float v, float* sred) {
#pragma unroll
    for (int o = 16; o; o >>= 1) v += __shfl_down(v, o, 32);
    __syncthreads();
    if ((threadIdx.x & 31) == 0) sred[threadIdx.x >> 5] = v;
    __syncthreads();
    float r = 0.f;
#pragma unroll
    for (int i = 0; i < 8; ++i) r += sred[i];
    return r;
}

__device__ inline float sigm(float x) { return 1.f / (1.f + expf(-x)); }

// ---------------- utility kernels ----------------
__global__ void __launch_bounds__(256) zero_f32_kernel(float* d, size_t n) {
    size_t i = (size_t)blockIdx.x * 256 + threadIdx.x;
    size_t st = (size_t)gridDim.x * 256;
    for (; i < n; i += st) d[i] = 0.f;
}

__global__ void __launch_bounds__(256) zero_f16_kernel(_Float16* d, size_t n) {
    size_t i = (size_t)blockIdx.x * 256 + threadIdx.x;
    size_t st = (size_t)gridDim.x * 256;
    for (; i < n; i += st) d[i] = (_Float16)0.f;
}

// fused transpose + f32->f16: dst[C][R] = src[R][C]^T ; R, C multiples of 32
__global__ void __launch_bounds__(256) transpose_f16_kernel(const float* __restrict__ src,
                                                            _Float16* __restrict__ dst,
                                                            int R, int C) {
    __shared__ float tile[32][33];
    int c0 = blockIdx.x * 32, r0 = blockIdx.y * 32;
    int tx = threadIdx.x & 31, ty = threadIdx.x >> 5;    // ty 0..7
#pragma unroll
    for (int i = 0; i < 4; ++i)
        tile[ty + i * 8][tx] = src[(size_t)(r0 + ty + i * 8) * C + c0 + tx];
    __syncthreads();
#pragma unroll
    for (int i = 0; i < 4; ++i)
        dst[(size_t)(c0 + ty + i * 8) * R + r0 + tx] = (_Float16)tile[tx][ty + i * 8];
}

// embedding gather: row r = t*B + b
__global__ void __launch_bounds__(256) embed_kernel(const int* __restrict__ tok,
                                                    const float* __restrict__ emb,
                                                    _Float16* __restrict__ x) {
    int r = blockIdx.x;
    int b = r & 31, t = r >> 5;
    int token = tok[b * T_ + t];
    const float* e = emb + (size_t)token * H_;
    for (int h = threadIdx.x; h < H_; h += 256)
        x[(size_t)r * H_ + h] = (_Float16)e[h];
}

// ---------------- big GEMM: C[M,N] = A[M,K] @ Bt[N,K]^T + bias[N] ----------------
// grid = (N/64, M/128), block = 256 (8 waves). Wave (mtp,nt): 4 M-tiles x 1 N-tile.
// 2x unrolled with two disjoint register sets, refilled IN PLACE right after use:
// WMMAs on set Y overlap the in-flight refill loads of set X (partial waitcnt, no copies).
__global__ void __launch_bounds__(256) gemm_big_kernel(const _Float16* __restrict__ A, int K,
                                                       const _Float16* __restrict__ Bt, int ldk,
                                                       int N, const float* __restrict__ bias,
                                                       float* __restrict__ C) {
    int tid = threadIdx.x, w = tid >> 5, lane = tid & 31;
    int nt = w & 3, mtp = w >> 2;
    int n0 = blockIdx.x * 64 + nt * 16;
    int m0 = blockIdx.y * 128 + mtp * 64;
    const _Float16* Bbase = Bt + (size_t)n0 * ldk;
    v8f acc[4] = {};
    int nch = K / KC;                       // 32 (K=1024), always even

    v16h bX, aX[4], bY, aY[4];
    bX = load_bT_frag(Bbase, ldk);
#pragma unroll
    for (int i = 0; i < 4; ++i) aX[i] = load_a_frag(A + (size_t)(m0 + i * 16) * K, K);
    bY = load_bT_frag(Bbase + KC, ldk);
#pragma unroll
    for (int i = 0; i < 4; ++i) aY[i] = load_a_frag(A + (size_t)(m0 + i * 16) * K + KC, K);

    for (int c = 0; c < nch; c += 2) {
#pragma unroll
        for (int i = 0; i < 4; ++i)
            acc[i] = __builtin_amdgcn_wmma_f32_16x16x32_f16(false, aX[i], false, bX, (short)0, acc[i], false, false);
        if (c + 2 < nch) {                  // refill set X with chunk c+2
            int k2 = (c + 2) * KC;
            bX = load_bT_frag(Bbase + k2, ldk);
#pragma unroll
            for (int i = 0; i < 4; ++i) aX[i] = load_a_frag(A + (size_t)(m0 + i * 16) * K + k2, K);
            __builtin_prefetch(Bbase + (size_t)(lane & 15) * ldk + k2 + 2 * KC, 0, 1);
        }
#pragma unroll
        for (int i = 0; i < 4; ++i)
            acc[i] = __builtin_amdgcn_wmma_f32_16x16x32_f16(false, aY[i], false, bY, (short)0, acc[i], false, false);
        if (c + 3 < nch) {                  // refill set Y with chunk c+3
            int k3 = (c + 3) * KC;
            bY = load_bT_frag(Bbase + k3, ldk);
#pragma unroll
            for (int i = 0; i < 4; ++i) aY[i] = load_a_frag(A + (size_t)(m0 + i * 16) * K + k3, K);
        }
    }

    int n  = n0 + (lane & 15);
    int mb = (lane >> 4) * 8;
    float bb = bias[n];
#pragma unroll
    for (int i = 0; i < 4; ++i) {
        int mrow = m0 + i * 16 + mb;
#pragma unroll
        for (int v = 0; v < 8; ++v)
            C[(size_t)(mrow + v) * N + n] = acc[i][v] + bb;
    }
}

// ---------------- persistent recurrent kernel: device-side phases ----------------
// GEMM phase: gates[32,4096] slice = A[32,K] @ Bt^T + add ; this block's slice = 64 columns.
// Even/odd accumulator chains double as the two register sets; in-place refill after use.
__device__ inline void gemm_phase(const _Float16* __restrict__ A, int K,
                                  const _Float16* __restrict__ Bt, int ldk,
                                  const float* __restrict__ add, int addStride,
                                  float* __restrict__ outg) {
    int tid = threadIdx.x, w = tid >> 5, lane = tid & 31;
    int mt = w >> 2, nt = w & 3;
    int n0 = blockIdx.x * 64 + nt * 16;
    const _Float16* Abase = A + (size_t)(mt * 16) * K;
    const _Float16* Bbase = Bt + (size_t)n0 * ldk;
    v8f acc0 = {}, acc1 = {};
    int nch = K / KC;                     // 32 or 64, always even

    v16h a0 = load_a_frag(Abase, K),      b0 = load_bT_frag(Bbase, ldk);
    v16h a1 = load_a_frag(Abase + KC, K), b1 = load_bT_frag(Bbase + KC, ldk);
    for (int c = 0; c < nch; c += 2) {
        acc0 = __builtin_amdgcn_wmma_f32_16x16x32_f16(false, a0, false, b0, (short)0, acc0, false, false);
        if (c + 2 < nch) {
            a0 = load_a_frag(Abase + (c + 2) * KC, K);
            b0 = load_bT_frag(Bbase + (c + 2) * KC, ldk);
        }
        acc1 = __builtin_amdgcn_wmma_f32_16x16x32_f16(false, a1, false, b1, (short)0, acc1, false, false);
        if (c + 3 < nch) {
            a1 = load_a_frag(Abase + (c + 3) * KC, K);
            b1 = load_bT_frag(Bbase + (c + 3) * KC, ldk);
        }
    }

    int n  = n0 + (lane & 15);
    int mb = (lane >> 4) * 8;
#pragma unroll
    for (int v = 0; v < 8; ++v) {
        int m = mt * 16 + mb + v;
        outg[(size_t)m * FOURH + n] = acc0[v] + acc1[v] + add[(size_t)m * addStride + n];
    }
}

// cell phase for batch row b (executed by blocks 0..31)
__device__ inline void cell_phase(int b,
                                  const float* __restrict__ gates, float* __restrict__ c,
                                  const float* __restrict__ lng, const float* __restrict__ lnb,
                                  _Float16* __restrict__ hA, int sA,
                                  _Float16* __restrict__ hB, int sB,
                                  float* sg, float* sred) {
    int tid = threadIdx.x;
    const float* g = gates + (size_t)b * 4096;
    for (int i = tid; i < 4096; i += 256) sg[i] = g[i];
    __syncthreads();

    float mean[4], rstd[4];
#pragma unroll
    for (int gi = 0; gi < 4; ++gi) {
        float s = 0.f;
        for (int j = tid; j < 1024; j += 256) s += sg[gi * 1024 + j];
        float m = block_sum(s, sred) * (1.f / 1024.f);
        float s2 = 0.f;
        for (int j = tid; j < 1024; j += 256) { float d = sg[gi * 1024 + j] - m; s2 += d * d; }
        float var = block_sum(s2, sred) * (1.f / 1024.f);
        mean[gi] = m; rstd[gi] = rsqrtf(var + EPS_);
    }

    float ncr[4], ovr[4];
#pragma unroll
    for (int k = 0; k < 4; ++k) {
        int j = tid + k * 256;
        float iv = (sg[j]        - mean[0]) * rstd[0] * lng[j]        + lnb[j];
        float jv = (sg[1024 + j] - mean[1]) * rstd[1] * lng[1024 + j] + lnb[1024 + j];
        float fv = (sg[2048 + j] - mean[2]) * rstd[2] * lng[2048 + j] + lnb[2048 + j];
        float ov = (sg[3072 + j] - mean[3]) * rstd[3] * lng[3072 + j] + lnb[3072 + j];
        float cp = c[(size_t)b * 1024 + j];
        float nc = cp * sigm(fv + 1.0f) + sigm(iv) * tanhf(jv);   // FORGET_BIAS = 1.0
        c[(size_t)b * 1024 + j] = nc;
        ncr[k] = nc; ovr[k] = ov;
    }

    float s = 0.f;
#pragma unroll
    for (int k = 0; k < 4; ++k) s += ncr[k];
    float mc = block_sum(s, sred) * (1.f / 1024.f);
    float s2 = 0.f;
#pragma unroll
    for (int k = 0; k < 4; ++k) { float d = ncr[k] - mc; s2 += d * d; }
    float vc = block_sum(s2, sred) * (1.f / 1024.f);
    float rc = rsqrtf(vc + EPS_);

#pragma unroll
    for (int k = 0; k < 4; ++k) {
        int j = tid + k * 256;
        float nh = tanhf((ncr[k] - mc) * rc * lng[4096 + j] + lnb[4096 + j]) * sigm(ovr[k]);
        _Float16 hh = (_Float16)nh;
        hA[(size_t)b * sA + j] = hh;
        hB[(size_t)b * sB + j] = hh;
    }
}

// device-wide barrier: monotone counter, one arrival per block per phase
__device__ inline void gsync(unsigned int* cnt, unsigned int& gen) {
    __threadfence();
    __syncthreads();
    gen += NBLK;
    if (threadIdx.x == 0) {
        atomicAdd(cnt, 1u);
        while (*((volatile unsigned int*)cnt) < gen)
            __builtin_amdgcn_s_sleep(2);
    }
    __syncthreads();
    __threadfence();
}

// persistent kernel: 128 timesteps x (gemm L0, cell L0, gemm L1, cell L1)
__global__ void __launch_bounds__(256) rnn_persistent_kernel(
        const _Float16* __restrict__ wT0,     // [4096][2048]  (W0^T; h-part at koff 1024)
        const _Float16* __restrict__ wT1,     // [4096][2048]  (W1^T)
        const float* __restrict__ g0in,       // [T*B][4096]   hoisted x@W0x + b0
        const float* __restrict__ b1,         // [4096]        layer-1 bias
        const float* __restrict__ lng, const float* __restrict__ lnb,
        float* __restrict__ gates,            // [32][4096]
        float* __restrict__ cst,              // [2][32][1024]
        _Float16* __restrict__ h0f,           // [32][1024]
        _Float16* __restrict__ hcat,          // [32][2048]
        _Float16* __restrict__ outs16,        // row b*T+t
        unsigned int* __restrict__ bar) {
    __shared__ float sg[4096];
    __shared__ float sred[8];
    unsigned int gen = 0;
    for (int t = 0; t < T_; ++t) {
        // layer 0 gates: h0_{t-1} @ W0h^T + g0in[t]
        gemm_phase(h0f, H_, wT0 + H_, 2 * H_, g0in + (size_t)t * B_ * FOURH, FOURH, gates);
        gsync(bar, gen);
        if (blockIdx.x < B_)
            cell_phase(blockIdx.x, gates, cst, lng, lnb, h0f, H_, hcat, 2 * H_, sg, sred);
        gsync(bar, gen);
        // layer 1 gates: [h0_t | h1_{t-1}] @ W1^T + b1
        gemm_phase(hcat, 2 * H_, wT1, 2 * H_, b1, 0, gates);
        gsync(bar, gen);
        if (blockIdx.x < B_)
            cell_phase(blockIdx.x, gates, cst + (size_t)B_ * H_, lng + 5 * H_, lnb + 5 * H_,
                       hcat + H_, 2 * H_, outs16 + (size_t)t * H_, T_ * H_, sg, sred);
        gsync(bar, gen);
    }
}

// ---------------- loss ----------------
__global__ void __launch_bounds__(256) row_loss_kernel(const float* __restrict__ logits,
                                                       const int* __restrict__ tgt,
                                                       float* __restrict__ rl) {
    __shared__ float sm[8], ss[8];
    int r = blockIdx.x, tid = threadIdx.x;
    const float* p = logits + (size_t)r * V_;
    float m = -3.4e38f, s = 0.f;
    for (int i = tid; i < V_; i += 256) {
        float x = p[i];
        float mn = fmaxf(m, x);
        s = s * expf(m - mn) + expf(x - mn);
        m = mn;
    }
#pragma unroll
    for (int o = 16; o; o >>= 1) {
        float m2 = __shfl_down(m, o, 32);
        float s2 = __shfl_down(s, o, 32);
        float mn = fmaxf(m, m2);
        s = s * expf(m - mn) + s2 * expf(m2 - mn);
        m = mn;
    }
    if ((tid & 31) == 0) { sm[tid >> 5] = m; ss[tid >> 5] = s; }
    __syncthreads();
    if (tid == 0) {
        float M = -3.4e38f;
#pragma unroll
        for (int i = 0; i < 8; ++i) M = fmaxf(M, sm[i]);
        float S = 0.f;
#pragma unroll
        for (int i = 0; i < 8; ++i) S += ss[i] * expf(sm[i] - M);
        rl[r] = (M + logf(S)) - p[tgt[r]];
    }
}

__global__ void __launch_bounds__(256) mean_kernel(const float* __restrict__ rl, float* __restrict__ out) {
    __shared__ float sred[8];
    float s = 0.f;
    for (int i = threadIdx.x; i < ROWS; i += 256) s += rl[i];
    s = block_sum(s, sred);
    if (threadIdx.x == 0) out[0] = s * (1.f / (float)ROWS);
}

// ---------------- host orchestration ----------------
extern "C" void kernel_launch(void* const* d_in, const int* in_sizes, int n_in,
                              void* d_out, int out_size, void* d_ws, size_t ws_size,
                              hipStream_t stream) {
    const int*   tok  = (const int*)d_in[0];
    const int*   tgt  = (const int*)d_in[1];
    const float* emb  = (const float*)d_in[2];
    const float* W    = (const float*)d_in[3];
    const float* brnn = (const float*)d_in[4];
    const float* lng  = (const float*)d_in[5];
    const float* lnb  = (const float*)d_in[6];
    const float* sw   = (const float*)d_in[7];
    const float* sb   = (const float*)d_in[8];
    float* out = (float*)d_out;

    char* wp = (char*)d_ws;
    auto alloc = [&](size_t bytes) { char* p = wp; wp += (bytes + 255) & ~(size_t)255; return p; };
    _Float16* x16    = (_Float16*)alloc((size_t)ROWS * H_ * 2);            // [T*B, H]
    _Float16* wT     = (_Float16*)alloc((size_t)2 * FOURH * 2048 * 2);     // W^T per layer
    _Float16* swT    = (_Float16*)alloc((size_t)V_ * H_ * 2);              // softmax_w^T
    _Float16* outs16 = (_Float16*)alloc((size_t)ROWS * H_ * 2);
    float*    g0in   = (float*)alloc((size_t)ROWS * FOURH * 4);
    float*    gates  = (float*)alloc((size_t)B_ * FOURH * 4);
    float*    cst    = (float*)alloc((size_t)2 * B_ * H_ * 4);
    _Float16* h0f    = (_Float16*)alloc((size_t)B_ * H_ * 2);
    _Float16* hcat   = (_Float16*)alloc((size_t)B_ * 2 * H_ * 2);
    float*    rloss  = (float*)alloc((size_t)ROWS * 4);
    unsigned int* bar = (unsigned int*)alloc(256);

    // pre-transpose weights to K-contiguous-per-column layout (once per call)
    transpose_f16_kernel<<<dim3(FOURH / 32, 2048 / 32), 256, 0, stream>>>(
        W, wT, 2048, FOURH);                                               // layer 0
    transpose_f16_kernel<<<dim3(FOURH / 32, 2048 / 32), 256, 0, stream>>>(
        W + (size_t)2048 * FOURH, wT + (size_t)FOURH * 2048, 2048, FOURH); // layer 1
    transpose_f16_kernel<<<dim3(V_ / 32, H_ / 32), 256, 0, stream>>>(
        sw, swT, H_, V_);

    zero_f32_kernel<<<64, 256, 0, stream>>>(cst, (size_t)2 * B_ * H_);
    zero_f16_kernel<<<64, 256, 0, stream>>>(h0f, (size_t)B_ * H_);
    zero_f16_kernel<<<64, 256, 0, stream>>>(hcat, (size_t)B_ * 2 * H_);
    zero_f32_kernel<<<1, 256, 0, stream>>>((float*)bar, 64);
    embed_kernel<<<ROWS, 256, 0, stream>>>(tok, emb, x16);

    // hoisted layer-0 input contribution: g0in = X @ W0x^T + b0
    gemm_big_kernel<<<dim3(FOURH / 64, ROWS / 128), 256, 0, stream>>>(
        x16, H_, wT, 2 * H_, FOURH, brnn, g0in);

    // full recurrence in ONE persistent cooperative kernel
    rnn_persistent_kernel<<<NBLK, 256, 0, stream>>>(
        wT, wT + (size_t)FOURH * 2048, g0in, brnn + FOURH,
        lng, lnb, gates, cst, h0f, hcat, outs16, bar);

    // logits = outs @ softmax_w + softmax_b
    gemm_big_kernel<<<dim3(V_ / 64, ROWS / 128), 256, 0, stream>>>(
        outs16, H_, swT, H_, V_, sb, out);

    row_loss_kernel<<<ROWS, 256, 0, stream>>>(out, tgt, rloss);
    mean_kernel<<<1, 256, 0, stream>>>(rloss, out + (size_t)ROWS * V_);
}
